// SwitchHeadAttention_1709396983886
// MI455X (gfx1250) — matrix operand order repaired and checked by reading, not compiled, not run on hardware
//
#include <hip/hip_runtime.h>
#include <hip/hip_bf16.h>
#include <math.h>

// ---------------------------------------------------------------------------
// CDNA5 (gfx1250) wave32 WMMA types
// ---------------------------------------------------------------------------
typedef __attribute__((ext_vector_type(16))) __bf16 v16bf;
typedef __attribute__((ext_vector_type(8)))  float  v8f;
typedef __attribute__((ext_vector_type(4)))  float  float4v;
typedef __attribute__((ext_vector_type(2)))  __bf16 bf16x2;
typedef __attribute__((ext_vector_type(4)))  __bf16 bf16x4;

#define WMMA_BF16(a, b, c) \
  __builtin_amdgcn_wmma_f32_16x16x32_bf16(false, (a), false, (b), (short)0, (c), false, false)

// A-fragment (16x32 bf16, MxK). Layout: lanes 0-15: M=lane, K=0..15;
// lanes 16-31: M=lane-16, K=16..31. lds is [m][k] with 'stride' elems/row.
__device__ __forceinline__ v16bf load_a_frag(const __bf16* lds, int stride, int lane) {
  const int m  = lane & 15;
  const int kb = (lane >> 4) << 4;      // 0 or 16
  const __bf16* row = lds + m * stride;
  v16bf a;
#pragma unroll
  for (int j = 0; j < 4; ++j) {
    bf16x2 p0 = *(const bf16x2*)(row + kb + 2 * j);        // K = kb+2j, kb+2j+1
    bf16x2 p1 = *(const bf16x2*)(row + 8 + kb + 2 * j);    // K = 8+kb+2j, ...
    a[2 * j]     = p0[0]; a[2 * j + 1]     = p0[1];
    a[8 + 2 * j] = p1[0]; a[8 + 2 * j + 1] = p1[1];
  }
  return a;
}

// B-fragment (32x16 bf16, KxN) read from LDS stored TRANSPOSED as [n][k].
// Contiguous-k layout used here (same as load_a_frag): lanes 0-15: N=lane, K=0..15; lanes 16-31: N=lane-16, K=16..31.
__device__ __forceinline__ v16bf load_b_frag(const __bf16* ldsT, int stride, int lane) {
  const int n  = lane & 15;
  const int kb = (lane >> 4) << 4;      // 0 or 16
  const __bf16* row = ldsT + n * stride;
  v16bf b;
#pragma unroll
  for (int j = 0; j < 8; ++j) {
    bf16x2 p = *(const bf16x2*)(row + kb + 2 * j);
    b[2 * j] = p[0]; b[2 * j + 1] = p[1];
  }
  return b;
}

// ---------------------------------------------------------------------------
// Vectorized 64x64 f32 -> bf16 tile stagers (128 threads; 8 x float4 each).
// Phase 1: issue all 8 global_load_b128 (loads stay in flight together).
// Phase 2: 2x v_cvt_pk_bf16_f32 + ds_store per chunk (partial loadcnt waits).
// ---------------------------------------------------------------------------
__device__ __forceinline__ bf16x4 cvt4(float4v f) {
  bf16x4 h;
  h[0] = (__bf16)f[0]; h[1] = (__bf16)f[1];
  h[2] = (__bf16)f[2]; h[3] = (__bf16)f[3];
  return h;
}

// direct copy: lds[r][c] = g[r*ldg + c]
__device__ __forceinline__ void tile_ld(__bf16 (*lds)[72], const float* __restrict__ g,
                                        int ldg, int tid) {
  float4v f[8];
#pragma unroll
  for (int it = 0; it < 8; ++it) {
    const int idx = (tid << 2) + (it << 9);
    const int r = idx >> 6, c = idx & 63;
    f[it] = *(const float4v*)(g + (size_t)r * ldg + c);
  }
#pragma unroll
  for (int it = 0; it < 8; ++it) {
    const int idx = (tid << 2) + (it << 9);
    const int r = idx >> 6, c = idx & 63;
    *(bf16x4*)&lds[r][c] = cvt4(f[it]);
  }
}

// transposed copy: ldsT[c][r] = g[r*ldg + c]  (coalesced reads, scattered b16 stores)
__device__ __forceinline__ void tile_ld_t(__bf16 (*ldsT)[72], const float* __restrict__ g,
                                          int ldg, int tid) {
  float4v f[8];
#pragma unroll
  for (int it = 0; it < 8; ++it) {
    const int idx = (tid << 2) + (it << 9);
    const int r = idx >> 6, c = idx & 63;
    f[it] = *(const float4v*)(g + (size_t)r * ldg + c);
  }
#pragma unroll
  for (int it = 0; it < 8; ++it) {
    const int idx = (tid << 2) + (it << 9);
    const int r = idx >> 6, c = idx & 63;
#pragma unroll
    for (int i = 0; i < 4; ++i) ldsT[c + i][r] = (__bf16)f[it][i];
  }
}

// ---------------------------------------------------------------------------
// Problem constants
// ---------------------------------------------------------------------------
#define BB   2
#define TT   1024
#define DIM  1024
#define HH   8
#define DHH  64
#define EE   5
#define KK   3
#define BT   (BB * TT)
#define SCALE 0.125f   // DH^-0.5

// ---------------------------------------------------------------------------
// Kernel 1: generic C[M,N] = A[M,K] @ W[K,N]  (fp32 in, bf16 WMMA, fp32 out)
// block = 128 threads (4 waves). 64x64 C tile. K stepped by 64 (8 WMMA / sync).
// ---------------------------------------------------------------------------
__global__ __launch_bounds__(128) void gemm_bf16(const float* __restrict__ A,
                                                 const float* __restrict__ W,
                                                 float* __restrict__ C,
                                                 int M, int N, int Kd) {
  __shared__ __align__(16) __bf16 lA[64][72];    // [m][k]
  __shared__ __align__(16) __bf16 lBT[64][72];   // [n][k]
  const int tid = threadIdx.x, lane = tid & 31, wave = tid >> 5;
  const int m0 = blockIdx.y * 64, n0 = blockIdx.x * 64;
  const bool nfull = (n0 + 64 <= N);

  v8f acc[4] = {v8f{0.f}, v8f{0.f}, v8f{0.f}, v8f{0.f}};

  for (int k0 = 0; k0 < Kd; k0 += 64) {
    __syncthreads();
    tile_ld(lA, A + (size_t)m0 * Kd + k0, Kd, tid);            // A tile [m][k]
    if (nfull) {
      tile_ld_t(lBT, W + (size_t)k0 * N + n0, N, tid);         // W tile -> [n][k]
    } else {                                                   // guarded path (N=40)
      for (int idx = tid; idx < 64 * 64; idx += 128) {
        int n = idx & 63, kk = idx >> 6;
        float w = (n0 + n < N) ? W[(size_t)(k0 + kk) * N + (n0 + n)] : 0.f;
        lBT[n][kk] = (__bf16)w;
      }
    }
    // prefetch next K tile into cache while WMMAs run
    if (k0 + 64 < Kd) {
      __builtin_prefetch(A + (size_t)(m0 + (tid & 63)) * Kd + k0 + 64, 0, 3);
      if (nfull) __builtin_prefetch(W + (size_t)(k0 + 64 + (tid & 63)) * N + n0, 0, 3);
    }
    __syncthreads();
#pragma unroll
    for (int ks = 0; ks < 64; ks += 32) {
      v16bf a = load_a_frag(&lA[wave * 16][ks], 72, lane);
#pragma unroll
      for (int s = 0; s < 4; ++s) {
        v16bf b = load_b_frag(&lBT[s * 16][ks], 72, lane);
        acc[s] = WMMA_BF16(a, b, acc[s]);
      }
    }
  }
  const int rbase = m0 + wave * 16 + ((lane >> 4) << 3);
  const int cl = lane & 15;
#pragma unroll
  for (int s = 0; s < 4; ++s) {
    int col = n0 + s * 16 + cl;
    if (col < N) {
#pragma unroll
      for (int r = 0; r < 8; ++r)
        C[(size_t)(rbase + r) * N + col] = acc[s][r];
    }
  }
}

// ---------------------------------------------------------------------------
// Kernel 2: routing — sigmoid + top-3 (soft gates for V-mix, 0/1 gates for out),
// and routed V mix into [B,H,T,DH]. One block per (bt,h), 64 threads over DH.
// ---------------------------------------------------------------------------
__global__ __launch_bounds__(64) void routing_kernel(const float* __restrict__ sscore,
                                                     const float* __restrict__ dscore,
                                                     const float* __restrict__ vexp,
                                                     float* __restrict__ vrouted,
                                                     float* __restrict__ sdgate) {
  const int bh = blockIdx.x;
  const int bt = bh >> 3, h = bh & 7;
  const int d = threadIdx.x;
  __shared__ float g[EE];
  if (d < EE) {
    const float* s  = sscore + (size_t)bt * (HH * EE) + h * EE;
    const float* sd = dscore + (size_t)bt * (HH * EE) + h * EE;
    float vi = 1.f / (1.f + __expf(-s[d]));
    float di = 1.f / (1.f + __expf(-sd[d]));
    int rs = 0, rd = 0;
#pragma unroll
    for (int j = 0; j < EE; ++j) {
      float vj = 1.f / (1.f + __expf(-s[j]));
      float dj = 1.f / (1.f + __expf(-sd[j]));
      rs += (vj > vi) || (vj == vi && j < d);
      rd += (dj > di) || (dj == di && j < d);
    }
    g[d] = (rs < KK) ? vi : 0.f;
    sdgate[(size_t)bt * (HH * EE) + h * EE + d] = (rd < KK) ? 1.f : 0.f;
  }
  __syncthreads();
  const float* ve = vexp + (size_t)bt * (HH * EE * DHH) + h * (EE * DHH) + d;
  float acc = 0.f;
#pragma unroll
  for (int e = 0; e < EE; ++e) acc += g[e] * ve[e * DHH];
  const int b = bt >> 10, t = bt & 1023;
  vrouted[(((size_t)b * HH + h) * TT + t) * DHH + d] = acc;
}

// ---------------------------------------------------------------------------
// Kernel 3: causal flash attention per (b,h). Block = one 64-row Q tile,
// 128 threads (4 waves, 16 rows each). Online softmax in f32, GEMMs in bf16 WMMA.
// ---------------------------------------------------------------------------
__global__ __launch_bounds__(128) void attention_kernel(const float* __restrict__ q,
                                                        const float* __restrict__ k,
                                                        const float* __restrict__ vr,
                                                        float* __restrict__ o) {
  __shared__ __align__(16) __bf16 lQ[64][72];    // [m][d]
  __shared__ __align__(16) __bf16 lK[64][72];    // [j][d]  (B^T layout for S=Q·K^T)
  __shared__ __align__(16) __bf16 lVT[64][72];   // [d][j]  (B^T layout for P·V)
  __shared__ __align__(16) __bf16 lP[64][72];    // [m][j]  (A layout for P·V)

  const int tid = threadIdx.x, lane = tid & 31, wave = tid >> 5;
  const int half = lane >> 4, cl = lane & 15;
  const int id = blockIdx.x;
  const int qt = id & 15, bh = id >> 4, h = bh & 7, b = bh >> 3;
  const int q0 = qt * 64;

  // load Q tile once (rows of q[b, q0+r, h, :])
  tile_ld(lQ, q + (size_t)(b * TT + q0) * (HH * DHH) + h * DHH, HH * DHH, tid);

  v8f   oacc[4] = {v8f{0.f}, v8f{0.f}, v8f{0.f}, v8f{0.f}};
  float mrow[8], lrow[8];
#pragma unroll
  for (int r = 0; r < 8; ++r) { mrow[r] = -INFINITY; lrow[r] = 0.f; }

  for (int j0 = 0; j0 <= q0; j0 += 64) {
    __syncthreads();
    tile_ld  (lK,  k  + (size_t)(b * TT + j0) * (HH * DHH) + h * DHH, HH * DHH, tid);
    tile_ld_t(lVT, vr + (((size_t)b * HH + h) * TT + j0) * DHH,       DHH,      tid);
    __syncthreads();

    // S = Q·K^T  (per wave: 16 rows x 64 cols)
    v8f sacc[4] = {v8f{0.f}, v8f{0.f}, v8f{0.f}, v8f{0.f}};
#pragma unroll
    for (int ks = 0; ks < 64; ks += 32) {
      v16bf a = load_a_frag(&lQ[wave * 16][ks], 72, lane);
#pragma unroll
      for (int s = 0; s < 4; ++s) {
        v16bf bfr = load_b_frag(&lK[s * 16][ks], 72, lane);
        sacc[s] = WMMA_BF16(a, bfr, sacc[s]);
      }
    }

    // scale + causal mask + tile row max
    const int rowg0 = q0 + wave * 16 + half * 8;   // + r
    float tmax[8];
#pragma unroll
    for (int r = 0; r < 8; ++r) tmax[r] = -INFINITY;
#pragma unroll
    for (int s = 0; s < 4; ++s) {
      int col = j0 + s * 16 + cl;
#pragma unroll
      for (int r = 0; r < 8; ++r) {
        float v = sacc[s][r] * SCALE;
        if (col > rowg0 + r) v = -1e30f;
        sacc[s][r] = v;
        tmax[r] = fmaxf(tmax[r], v);
      }
    }
#pragma unroll
    for (int m = 8; m >= 1; m >>= 1)
#pragma unroll
      for (int r = 0; r < 8; ++r)
        tmax[r] = fmaxf(tmax[r], __shfl_xor(tmax[r], m, 32));

    // online softmax rescale
    float rsum[8];
#pragma unroll
    for (int r = 0; r < 8; ++r) {
      float mn = fmaxf(mrow[r], tmax[r]);
      float alpha = __expf(mrow[r] - mn);
      mrow[r] = mn;
      lrow[r] *= alpha;
#pragma unroll
      for (int s = 0; s < 4; ++s) oacc[s][r] *= alpha;
      rsum[r] = 0.f;
    }
    // P = exp(S - m); stash bf16 P in per-wave LDS slice (C-layout -> A-layout)
#pragma unroll
    for (int s = 0; s < 4; ++s)
#pragma unroll
      for (int r = 0; r < 8; ++r) {
        float p = __expf(sacc[s][r] - mrow[r]);
        rsum[r] += p;
        lP[wave * 16 + half * 8 + r][s * 16 + cl] = (__bf16)p;
      }
#pragma unroll
    for (int m = 8; m >= 1; m >>= 1)
#pragma unroll
      for (int r = 0; r < 8; ++r) rsum[r] += __shfl_xor(rsum[r], m, 32);
#pragma unroll
    for (int r = 0; r < 8; ++r) lrow[r] += rsum[r];

    // O += P·V  (same-wave LDS RAW is in-order)
#pragma unroll
    for (int ks = 0; ks < 64; ks += 32) {
      v16bf a = load_a_frag(&lP[wave * 16][ks], 72, lane);
#pragma unroll
      for (int s = 0; s < 4; ++s) {
        v16bf bfr = load_b_frag(&lVT[s * 16][ks], 72, lane);
        oacc[s] = WMMA_BF16(a, bfr, oacc[s]);
      }
    }
  }

  // epilogue: normalize and store o[b,h,t,d]
#pragma unroll
  for (int s = 0; s < 4; ++s) {
    int col = s * 16 + cl;
#pragma unroll
    for (int r = 0; r < 8; ++r) {
      int row = q0 + wave * 16 + half * 8 + r;
      o[(((size_t)b * HH + h) * TT + row) * DHH + col] = oacc[s][r] / lrow[r];
    }
  }
}

// ---------------------------------------------------------------------------
// Kernel 4: out[bt,d] = sum_{h,e} sd[bt,h,e] * o[b,h,t,:] · Wo[h,e,d,:]
// 64x64 output tile per block; inner loop over 40 (h,e) experts, K=64 each.
// ---------------------------------------------------------------------------
__global__ __launch_bounds__(128) void out_proj_kernel(const float* __restrict__ o,
                                                       const float* __restrict__ sdgate,
                                                       const float* __restrict__ Wo,
                                                       float* __restrict__ out) {
  __shared__ __align__(16) __bf16 lA[64][72];    // [m][c] gated o rows
  __shared__ __align__(16) __bf16 lBT[64][72];   // [d][c] = Wo[h,e,d,c] (already [n][k])
  const int tid = threadIdx.x, lane = tid & 31, wave = tid >> 5;
  const int m0 = blockIdx.y * 64, d0 = blockIdx.x * 64;

  v8f acc[4] = {v8f{0.f}, v8f{0.f}, v8f{0.f}, v8f{0.f}};

  for (int he = 0; he < HH * EE; ++he) {
    const int h = he / EE, e = he % EE;
    __syncthreads();
    // gated A tile: lA[r][c] = sd[m0+r,h,e] * o[b,h,t,c]  (batched loads)
    float4v f[8];
    float   gt[8];
#pragma unroll
    for (int it = 0; it < 8; ++it) {
      const int idx = (tid << 2) + (it << 9);
      const int r = idx >> 6, c = idx & 63;
      const int bt = m0 + r, b = bt >> 10, t = bt & 1023;
      gt[it] = sdgate[(size_t)bt * (HH * EE) + h * EE + e];
      f[it]  = *(const float4v*)(o + (((size_t)b * HH + h) * TT + t) * DHH + c);
    }
#pragma unroll
    for (int it = 0; it < 8; ++it) {
      const int idx = (tid << 2) + (it << 9);
      const int r = idx >> 6, c = idx & 63;
      float4v v = f[it];
      v[0] *= gt[it]; v[1] *= gt[it]; v[2] *= gt[it]; v[3] *= gt[it];
      *(bf16x4*)&lA[r][c] = cvt4(v);
    }
    // Wo tile: [d][c] is already B^T ([n][k]) layout — direct vectorized copy
    tile_ld(lBT, Wo + ((size_t)h * EE + e) * DIM * DHH + (size_t)d0 * DHH, DHH, tid);
    __syncthreads();
#pragma unroll
    for (int ks = 0; ks < 64; ks += 32) {
      v16bf a = load_a_frag(&lA[wave * 16][ks], 72, lane);
#pragma unroll
      for (int s = 0; s < 4; ++s) {
        v16bf bfr = load_b_frag(&lBT[s * 16][ks], 72, lane);
        acc[s] = WMMA_BF16(a, bfr, acc[s]);
      }
    }
  }
  const int rbase = m0 + wave * 16 + ((lane >> 4) << 3);
  const int cl = lane & 15;
#pragma unroll
  for (int s = 0; s < 4; ++s) {
    int col = d0 + s * 16 + cl;
#pragma unroll
    for (int r = 0; r < 8; ++r)
      out[(size_t)(rbase + r) * DIM + col] = acc[s][r];
  }
}

// ---------------------------------------------------------------------------
// Host launcher
// ---------------------------------------------------------------------------
extern "C" void kernel_launch(void* const* d_in, const int* in_sizes, int n_in,
                              void* d_out, int out_size, void* d_ws, size_t ws_size,
                              hipStream_t stream) {
  const float* x  = (const float*)d_in[0];
  const float* Wq = (const float*)d_in[1];
  const float* Wk = (const float*)d_in[2];
  const float* Wv = (const float*)d_in[3];
  const float* Ws = (const float*)d_in[4];
  const float* Wd = (const float*)d_in[5];
  const float* Wo = (const float*)d_in[6];
  float* out = (float*)d_out;

  float* ws = (float*)d_ws;
  size_t off = 0;
  float* qb     = ws + off; off += (size_t)BT * HH * DHH;        // 2048*512
  float* kb     = ws + off; off += (size_t)BT * HH * DHH;
  float* vexp   = ws + off; off += (size_t)BT * HH * EE * DHH;   // 2048*2560
  float* sscore = ws + off; off += (size_t)BT * HH * EE;
  float* dscore = ws + off; off += (size_t)BT * HH * EE;
  float* sdg    = ws + off; off += (size_t)BT * HH * EE;
  float* vr     = ws + off; off += (size_t)BB * HH * TT * DHH;
  float* ob     = ws + off; off += (size_t)BB * HH * TT * DHH;

  const dim3 blk(128);
  // projections: x[2048,1024] @ W
  gemm_bf16<<<dim3(HH * DHH / 64, BT / 64),        blk, 0, stream>>>(x, Wq, qb,     BT, HH * DHH,      DIM);
  gemm_bf16<<<dim3(HH * DHH / 64, BT / 64),        blk, 0, stream>>>(x, Wk, kb,     BT, HH * DHH,      DIM);
  gemm_bf16<<<dim3(HH * EE * DHH / 64, BT / 64),   blk, 0, stream>>>(x, Wv, vexp,   BT, HH * EE * DHH, DIM);
  gemm_bf16<<<dim3(1, BT / 64),                    blk, 0, stream>>>(x, Ws, sscore, BT, HH * EE,       DIM);
  gemm_bf16<<<dim3(1, BT / 64),                    blk, 0, stream>>>(x, Wd, dscore, BT, HH * EE,       DIM);
  // routing + V mix
  routing_kernel<<<dim3(BT * HH), dim3(64), 0, stream>>>(sscore, dscore, vexp, vr, sdg);
  // causal attention
  attention_kernel<<<dim3(BB * HH * (TT / 64)), blk, 0, stream>>>(qb, kb, vr, ob);
  // gated expert output projection
  out_proj_kernel<<<dim3(DIM / 64, BT / 64), blk, 0, stream>>>(ob, sdg, Wo, out);
}
